// LiquidReasoningTransformer_44882408243566
// MI455X (gfx1250) — matrix operand order repaired and checked by reading, not compiled, not run on hardware
//
#include <hip/hip_runtime.h>
#include <hip/hip_bf16.h>

// ---------------------------------------------------------------------------
// LiquidReasoningTransformer for MI455X (gfx1250, wave32, WMMA)
// Bandwidth-bound: bf16 weights (302MB) stream from HBM each of 8 steps.
// GEMMs: v_wmma_f32_16x16x32_bf16, weights pre-transposed to [N][K] bf16,
// double-buffered K loop, split-K x2 per block (LDS combine), fused QKV.
// ---------------------------------------------------------------------------

#define HDIM 1024
#define FF   4096
#define NH   16
#define HD   64
#define SEQ  65     // 64 board tokens + 1 reasoning token
#define MPAD 80     // SEQ padded to 5 tiles of 16
#define MT   5      // M tiles per wave
#define NL   12
#define PDIM 4672
#define FEAT 112
#define QKVN 3072   // fused q|k|v output width
#define MAX_STEPS 8 // setup_inputs() fixes max_steps = 8 (device scalar, unreadable in capture)

typedef __bf16 v16bf __attribute__((ext_vector_type(16)));
typedef __bf16 v8bf  __attribute__((ext_vector_type(8)));
typedef float  v8f   __attribute__((ext_vector_type(8)));

union BF16x16 { v16bf v; v8bf h[2]; };

__device__ __forceinline__ unsigned short f2bf(float f) {
  unsigned int u = __float_as_uint(f);
  unsigned int r = u + 0x7FFFu + ((u >> 16) & 1u);   // round-to-nearest-even
  return (unsigned short)(r >> 16);
}

// ---------------------------------------------------------------------------
// fp32 [K x N] row-major  ->  bf16 [N x K] (transposed, so WMMA B-fragments
// are contiguous per-lane loads). Tiled LDS transpose for coalescing.
// ---------------------------------------------------------------------------
__global__ void transpose_to_bf16_kernel(const float* __restrict__ W,
                                         unsigned short* __restrict__ Wt,
                                         int K, int N) {
  __shared__ float tile[32][33];
  int kb = blockIdx.y * 32, nb = blockIdx.x * 32;
  int tx = threadIdx.x, ty = threadIdx.y;     // 32 x 8
  for (int i = 0; i < 32; i += 8) {
    int k = kb + ty + i, n = nb + tx;
    tile[ty + i][tx] = (k < K && n < N) ? W[(size_t)k * N + n] : 0.f;
  }
  __syncthreads();
  for (int i = 0; i < 32; i += 8) {
    int n = nb + ty + i, k = kb + tx;
    if (n < N && k < K) Wt[(size_t)n * K + k] = f2bf(tile[tx][ty + i]);
  }
}

// ---------------------------------------------------------------------------
// Fragment load helpers (wave32, contiguous k order shared by A and B, not the ISA table order):
//   A: lane L holds row M=L&15; lanes 0-15 K=[0..15],
//      lanes 16-31 K=[16..31]  -> two 16B loads per tile.
//   B: lane L holds col N=L&15; lanes 0-15 K=[0..15], lanes 16-31 K=[16..31]
//      -> one contiguous 32B load from the transposed weights.
// ---------------------------------------------------------------------------
__device__ __forceinline__ void load_frag(const __bf16* __restrict__ W, size_t brow,
                                          const __bf16* __restrict__ X,
                                          int l16, int hi, int K, int k0,
                                          BF16x16& b, BF16x16 (&a)[MT]) {
  b.v = *(const v16bf*)(W + brow + k0 + hi * 16);
#pragma unroll
  for (int m = 0; m < MT; ++m) {
    const __bf16* ap = X + (size_t)(m * 16 + l16) * (size_t)K + k0 + hi * 16;
    a[m].h[0] = *(const v8bf*)(ap);
    a[m].h[1] = *(const v8bf*)(ap + 8);
  }
}

// ---------------------------------------------------------------------------
// WMMA GEMM:  Y[M x N] = Xb[MPAD x K](bf16) @ Wt[N x K](bf16, transposed)
//             + bias, optional residual, optional relu + adaptive-width mask
//             (mask width recomputed from row-std stats in the epilogue).
// Block = 64 threads = 2 waves: one 16-column stripe, split-K x2, LDS combine.
// K loop is double-buffered so WMMAs overlap the next chunk's loads.
// ---------------------------------------------------------------------------
template<bool RELU_MASK, bool RESID>
__launch_bounds__(64)
__global__ void wmma_gemm_kernel(const unsigned short* __restrict__ Xb,
                                 const unsigned short* __restrict__ Wtw,
                                 const float* __restrict__ bias,
                                 const float* __restrict__ resid,
                                 const float* __restrict__ stats,
                                 float* __restrict__ Yf,
                                 unsigned short* __restrict__ Yb,
                                 int N, int K, int M) {
  __shared__ float part[MT * 8 * 32];
  const int wv   = threadIdx.x >> 5;
  const int lane = threadIdx.x & 31;
  const int col0 = blockIdx.x * 16;
  const int l16  = lane & 15;
  const int hi   = lane >> 4;
  const int khalf = K >> 1;                 // K is 1024 or 4096: even # of 64-chunks
  const int kbeg  = wv * khalf;
  const int kend  = kbeg + khalf;

  v8f acc[MT];
  const v8f vz = {0.f, 0.f, 0.f, 0.f, 0.f, 0.f, 0.f, 0.f};
#pragma unroll
  for (int m = 0; m < MT; ++m) acc[m] = vz;

  const __bf16* X = (const __bf16*)Xb;
  const __bf16* W = (const __bf16*)Wtw;
  const size_t brow = (size_t)(col0 + l16) * (size_t)K;

  BF16x16 b0, b1;
  BF16x16 a0[MT], a1[MT];
  load_frag(W, brow, X, l16, hi, K, kbeg, b0, a0);

  for (int k0 = kbeg; k0 < kend; k0 += 64) {
    __builtin_prefetch(W + brow + k0 + 128, 0, 1);       // global_prefetch_b8
    load_frag(W, brow, X, l16, hi, K, k0 + 32, b1, a1);  // fill buffer 1
#pragma unroll
    for (int m = 0; m < MT; ++m)
      acc[m] = __builtin_amdgcn_wmma_f32_16x16x32_bf16(
          false, a0[m].v, false, b0.v, (short)0, acc[m], false, false);
    if (k0 + 64 < kend)
      load_frag(W, brow, X, l16, hi, K, k0 + 64, b0, a0);  // refill buffer 0
#pragma unroll
    for (int m = 0; m < MT; ++m)
      acc[m] = __builtin_amdgcn_wmma_f32_16x16x32_bf16(
          false, a1[m].v, false, b1.v, (short)0, acc[m], false, false);
  }

  // split-K combine: wave 1 parks partials in LDS, wave 0 reduces + epilogue
  if (wv == 1) {
#pragma unroll
    for (int m = 0; m < MT; ++m)
#pragma unroll
      for (int r = 0; r < 8; ++r) part[m * 256 + r * 32 + lane] = acc[m][r];
  }
  __syncthreads();
  if (wv == 0) {
    float aw = 0.f;
    if (RELU_MASK) {   // confusion -> adaptive FF width, folded into epilogue
      float s = 0.f;
      for (int i = 0; i < SEQ; ++i) s += stats[i];
      aw = floorf((float)FF * (0.5f + 0.5f * tanhf((s * (1.f / SEQ)) * 10.f)));
    }
    const int   col = col0 + l16;
    const float bc  = bias[col];
#pragma unroll
    for (int m = 0; m < MT; ++m) {
#pragma unroll
      for (int r = 0; r < 8; ++r) {
        int row = m * 16 + r + hi * 8;
        if (row < M) {
          float y = acc[m][r] + part[m * 256 + r * 32 + lane] + bc;
          if (RELU_MASK) y = (y > 0.f && (float)col < aw) ? y : 0.f;
          if (RESID)     y += resid[(size_t)row * N + col];
          if (Yf) Yf[(size_t)row * N + col] = y;
          if (Yb) Yb[(size_t)row * N + col] = f2bf(y);
        }
      }
    }
  }
}

// ---------------------------------------------------------------------------
// board_emb = board @ We + be        (64x112 @ 112x1024 — tiny)
// ---------------------------------------------------------------------------
__global__ void embed_kernel(const float* __restrict__ board,
                             const float* __restrict__ We,
                             const float* __restrict__ be,
                             float* __restrict__ emb) {
  int idx = blockIdx.x * blockDim.x + threadIdx.x;
  if (idx >= 64 * HDIM) return;
  int r = idx / HDIM, c = idx % HDIM;
  float a = be[c];
  for (int t = 0; t < FEAT; ++t) a += board[r * FEAT + t] * We[(size_t)t * HDIM + c];
  emb[idx] = a;
}

// tokens = concat(board_emb, token): rows 0..63 = emb, row 64 = token (f32+bf16)
__global__ void concat_tokens_kernel(const float* __restrict__ emb,
                                     const float* __restrict__ token,
                                     float* __restrict__ xf,
                                     unsigned short* __restrict__ xb) {
  int idx = blockIdx.x * blockDim.x + threadIdx.x;
  if (idx >= SEQ * HDIM) return;
  int row = idx / HDIM, col = idx % HDIM;
  float v = (row < 64) ? emb[idx] : token[col];
  xf[idx] = v;
  xb[idx] = f2bf(v);
}

// ---------------------------------------------------------------------------
// Attention over the fused QKV buffer (row stride 3072): one block per head.
// ---------------------------------------------------------------------------
__global__ void attention_kernel(const float* __restrict__ QKV,
                                 unsigned short* __restrict__ Ob) {
  __shared__ float sc[SEQ][SEQ];
  const int h = blockIdx.x;
  const int tid = threadIdx.x;
  const float scale = 0.125f;   // 1/sqrt(64)
  for (int idx = tid; idx < SEQ * SEQ; idx += blockDim.x) {
    int qi = idx / SEQ, ki = idx % SEQ;
    const float* qp = QKV + (size_t)qi * QKVN + h * HD;
    const float* kp = QKV + (size_t)ki * QKVN + HDIM + h * HD;
    float d = 0.f;
    for (int t = 0; t < HD; ++t) d += qp[t] * kp[t];
    sc[qi][ki] = d * scale;
  }
  __syncthreads();
  for (int qi = tid; qi < SEQ; qi += blockDim.x) {
    float mx = -3.4e38f;
    for (int k = 0; k < SEQ; ++k) mx = fmaxf(mx, sc[qi][k]);
    float s = 0.f;
    for (int k = 0; k < SEQ; ++k) { float e = __expf(sc[qi][k] - mx); sc[qi][k] = e; s += e; }
    float inv = 1.f / s;
    for (int k = 0; k < SEQ; ++k) sc[qi][k] *= inv;
  }
  __syncthreads();
  for (int idx = tid; idx < SEQ * HD; idx += blockDim.x) {
    int qi = idx / HD, d = idx % HD;
    float o = 0.f;
    for (int ki = 0; ki < SEQ; ++ki)
      o += sc[qi][ki] * QKV[(size_t)ki * QKVN + 2 * HDIM + h * HD + d];
    Ob[(size_t)qi * HDIM + h * HD + d] = f2bf(o);
  }
}

// ---------------------------------------------------------------------------
// LayerNorm per row (block per row); optionally emits per-row std of output.
// ---------------------------------------------------------------------------
template<bool STD>
__global__ void layernorm_kernel(const float* __restrict__ in,
                                 const float* __restrict__ s,
                                 const float* __restrict__ b,
                                 float* __restrict__ xf,
                                 unsigned short* __restrict__ xb,
                                 float* __restrict__ stats) {
  __shared__ float rs[256], rs2[256];
  const int row = blockIdx.x, tid = threadIdx.x;
  const float* xr = in + (size_t)row * HDIM;
  float a = 0.f, a2 = 0.f;
  for (int i = tid; i < HDIM; i += 256) { float v = xr[i]; a += v; a2 += v * v; }
  rs[tid] = a; rs2[tid] = a2; __syncthreads();
  for (int w = 128; w > 0; w >>= 1) {
    if (tid < w) { rs[tid] += rs[tid + w]; rs2[tid] += rs2[tid + w]; }
    __syncthreads();
  }
  const float mu  = rs[0] * (1.f / HDIM);
  const float var = rs2[0] * (1.f / HDIM) - mu * mu;
  const float inv = rsqrtf(var + 1e-6f);
  __syncthreads();
  float ya = 0.f, ya2 = 0.f;
  for (int i = tid; i < HDIM; i += 256) {
    float y = (xr[i] - mu) * inv * s[i] + b[i];
    xf[(size_t)row * HDIM + i] = y;
    xb[(size_t)row * HDIM + i] = f2bf(y);
    ya += y; ya2 += y * y;
  }
  if (STD) {
    rs[tid] = ya; rs2[tid] = ya2; __syncthreads();
    for (int w = 128; w > 0; w >>= 1) {
      if (tid < w) { rs[tid] += rs[tid + w]; rs2[tid] += rs2[tid + w]; }
      __syncthreads();
    }
    if (tid == 0) {
      float m = rs[0] * (1.f / HDIM);
      float v = rs2[0] * (1.f / HDIM) - m * m;
      stats[row] = sqrtf(fmaxf(v, 0.f));
    }
  }
}

// ---------------------------------------------------------------------------
// Step gating: keep/stop probabilities, conditional token update.
// st[1] = active flag, st[2] = step count.
// ---------------------------------------------------------------------------
__global__ void gate_kernel(const float* __restrict__ xf, float* __restrict__ token,
                            const float* __restrict__ dw, const float* __restrict__ db,
                            const float* __restrict__ sw, const float* __restrict__ sb,
                            float* __restrict__ st) {
  __shared__ float red[256];
  __shared__ float keep_s, stop_s;
  const int tid = threadIdx.x;
  const float* nt = xf + (size_t)64 * HDIM;
  float d = 0.f;
  for (int i = tid; i < HDIM; i += 256) d += token[i] * dw[i] + nt[i] * dw[HDIM + i];
  red[tid] = d; __syncthreads();
  for (int w = 128; w > 0; w >>= 1) { if (tid < w) red[tid] += red[tid + w]; __syncthreads(); }
  if (tid == 0) keep_s = 1.f / (1.f + __expf(-(red[0] + db[0])));
  __syncthreads();
  const float keep = keep_s;
  float sdot = 0.f;
  for (int i = tid; i < HDIM; i += 256) {
    float cand = (keep > 0.5f) ? nt[i] : token[i];
    sdot += cand * sw[i];
  }
  __syncthreads();
  red[tid] = sdot; __syncthreads();
  for (int w = 128; w > 0; w >>= 1) { if (tid < w) red[tid] += red[tid + w]; __syncthreads(); }
  if (tid == 0) stop_s = 1.f / (1.f + __expf(-(red[0] + sb[0])));
  __syncthreads();
  const float active = st[1];
  if (active > 0.5f && keep > 0.5f) {
    for (int i = tid; i < HDIM; i += 256) token[i] = nt[i];
  }
  if (tid == 0 && active > 0.5f) {
    st[2] += 1.f;
    if (stop_s >= 0.95f) st[1] = 0.f;
  }
}

__global__ void reset_state_kernel(float* st) {
  if (threadIdx.x == 0) { st[0] = 0.f; st[1] = 1.f; st[2] = 0.f; }
}

__global__ void board_mean_kernel(const float* __restrict__ emb, float* __restrict__ bm) {
  int c = blockIdx.x * blockDim.x + threadIdx.x;
  if (c >= HDIM) return;
  float a = 0.f;
  for (int r = 0; r < 64; ++r) a += emb[(size_t)r * HDIM + c];
  bm[c] = a * (1.f / 64.f);
}

// policy logits + legality (two 2048x4672 GEMVs, memory-bound, coalesced in j)
__global__ void head_logits_kernel(const float* __restrict__ token,
                                   const float* __restrict__ bmean,
                                   const float* __restrict__ pw, const float* __restrict__ pb,
                                   const float* __restrict__ cw, const float* __restrict__ cb,
                                   float* __restrict__ logits, float* __restrict__ legal) {
  __shared__ float cat[2 * HDIM];
  const int tid = threadIdx.x;
  for (int i = tid; i < 2 * HDIM; i += blockDim.x)
    cat[i] = (i < HDIM) ? token[i] : bmean[i - HDIM];
  __syncthreads();
  int j = blockIdx.x * blockDim.x + tid;
  if (j >= PDIM) return;
  float lp = pb[j], lc = cb[j];
  for (int i = 0; i < 2 * HDIM; ++i) {
    float c = cat[i];
    lp += c * pw[(size_t)i * PDIM + j];
    lc += c * cw[(size_t)i * PDIM + j];
  }
  logits[j] = lp;
  legal[j]  = 1.f / (1.f + __expf(-lc));
}

// value head + softmax(policy)*legal + steps  ->  d_out [1 + 4672 + 1]
__global__ void final_kernel(const float* __restrict__ token,
                             const float* __restrict__ vw, const float* __restrict__ vb,
                             const float* __restrict__ logits, const float* __restrict__ legal,
                             const float* __restrict__ st, float* __restrict__ out) {
  __shared__ float red[512];
  __shared__ float sval, smax, ssum;
  const int tid = threadIdx.x;
  float v = 0.f;
  for (int i = tid; i < HDIM; i += 512) v += token[i] * vw[i];
  red[tid] = v; __syncthreads();
  for (int w = 256; w > 0; w >>= 1) { if (tid < w) red[tid] += red[tid + w]; __syncthreads(); }
  if (tid == 0) sval = red[0] + vb[0];
  __syncthreads();
  float mx = -3.4e38f;
  for (int j = tid; j < PDIM; j += 512) mx = fmaxf(mx, logits[j]);
  red[tid] = mx; __syncthreads();
  for (int w = 256; w > 0; w >>= 1) { if (tid < w) red[tid] = fmaxf(red[tid], red[tid + w]); __syncthreads(); }
  if (tid == 0) smax = red[0];
  __syncthreads();
  float se = 0.f;
  for (int j = tid; j < PDIM; j += 512) se += __expf(logits[j] - smax);
  red[tid] = se; __syncthreads();
  for (int w = 256; w > 0; w >>= 1) { if (tid < w) red[tid] += red[tid + w]; __syncthreads(); }
  if (tid == 0) ssum = red[0];
  __syncthreads();
  const float inv = 1.f / ssum;
  for (int j = tid; j < PDIM; j += 512)
    out[1 + j] = __expf(logits[j] - smax) * inv * legal[j];
  if (tid == 0) { out[0] = sval; out[1 + PDIM] = st[2]; }
}

// ---------------------------------------------------------------------------
// Host orchestration
// ---------------------------------------------------------------------------
extern "C" void kernel_launch(void* const* d_in, const int* in_sizes, int n_in,
                              void* d_out, int out_size, void* d_ws, size_t ws_size,
                              hipStream_t stream) {
  (void)in_sizes; (void)n_in; (void)out_size; (void)ws_size;

  const float* board = (const float*)d_in[0];
  const float* rtok  = (const float*)d_in[1];
  const float* embW  = (const float*)d_in[2];
  const float* embB  = (const float*)d_in[3];

  struct Layer {
    const float *wq, *bq, *wk, *bk, *wv, *bv, *wo, *bo;
    const float *ln1s, *ln1b, *ff1, *b1, *ff2, *b2, *ln2s, *ln2b;
  };
  Layer L[NL];
  for (int l = 0; l < NL; ++l) {
    const float* const* p = (const float* const*)(d_in + 4 + l * 16);
    L[l] = { p[0], p[1], p[2], p[3], p[4], p[5], p[6], p[7],
             p[8], p[9], p[10], p[11], p[12], p[13], p[14], p[15] };
  }
  const int hb_ = 4 + NL * 16;
  const float* dw  = (const float*)d_in[hb_ + 0];
  const float* db  = (const float*)d_in[hb_ + 1];
  const float* sw  = (const float*)d_in[hb_ + 2];
  const float* sb  = (const float*)d_in[hb_ + 3];
  const float* cw  = (const float*)d_in[hb_ + 4];
  const float* cb  = (const float*)d_in[hb_ + 5];
  const float* vw  = (const float*)d_in[hb_ + 6];
  const float* vb  = (const float*)d_in[hb_ + 7];
  const float* pw  = (const float*)d_in[hb_ + 8];
  const float* pb  = (const float*)d_in[hb_ + 9];

  // ---- workspace bump allocator -------------------------------------------
  char* ws = (char*)d_ws;
  size_t off = 0;
  auto alloc = [&](size_t bytes) -> void* {
    off = (off + 255) & ~(size_t)255;
    void* p = ws + off;
    off += bytes;
    return p;
  };

  unsigned short* Wqkv[NL];   // [3072][1024] bf16 (q|k|v stacked on N)
  unsigned short* Wo[NL];     // [1024][1024]
  unsigned short* Wf1[NL];    // [4096][1024]
  unsigned short* Wf2[NL];    // [1024][4096]
  float*          Bqkv[NL];   // packed q|k|v bias [3072]
  for (int l = 0; l < NL; ++l) {
    Wqkv[l] = (unsigned short*)alloc((size_t)QKVN * HDIM * 2);
    Wo[l]   = (unsigned short*)alloc((size_t)HDIM * HDIM * 2);
    Wf1[l]  = (unsigned short*)alloc((size_t)FF * HDIM * 2);
    Wf2[l]  = (unsigned short*)alloc((size_t)HDIM * FF * 2);
    Bqkv[l] = (float*)alloc((size_t)QKVN * 4);
  }
  off = (off + 255) & ~(size_t)255;
  const size_t actBegin = off;
  float*          emb    = (float*)alloc((size_t)64 * HDIM * 4);
  float*          token  = (float*)alloc((size_t)HDIM * 4);
  float*          xf     = (float*)alloc((size_t)MPAD * HDIM * 4);
  unsigned short* xb     = (unsigned short*)alloc((size_t)MPAD * HDIM * 2);
  float*          yqkv   = (float*)alloc((size_t)MPAD * QKVN * 4);
  unsigned short* ob     = (unsigned short*)alloc((size_t)MPAD * HDIM * 2);
  float*          tmp    = (float*)alloc((size_t)MPAD * HDIM * 4);
  unsigned short* hb     = (unsigned short*)alloc((size_t)MPAD * FF * 2);
  float*          stats  = (float*)alloc((size_t)MPAD * 4);
  float*          st     = (float*)alloc(16 * 4);
  float*          bm     = (float*)alloc((size_t)HDIM * 4);
  float*          logits = (float*)alloc((size_t)PDIM * 4);
  float*          legal  = (float*)alloc((size_t)PDIM * 4);
  const size_t actEnd = off;

  // zero activation region (pad rows of bf16 buffers must be 0 for WMMA)
  hipMemsetAsync(ws + actBegin, 0, actEnd - actBegin, stream);
  reset_state_kernel<<<1, 32, 0, stream>>>(st);

  // ---- one-time per call: fp32 -> transposed bf16 weights, packed QKV -----
  const dim3 tb(32, 8);
  const dim3 gHH(HDIM / 32, HDIM / 32);
  for (int l = 0; l < NL; ++l) {
    transpose_to_bf16_kernel<<<gHH, tb, 0, stream>>>(L[l].wq, Wqkv[l] + 0 * HDIM * HDIM, HDIM, HDIM);
    transpose_to_bf16_kernel<<<gHH, tb, 0, stream>>>(L[l].wk, Wqkv[l] + 1 * HDIM * HDIM, HDIM, HDIM);
    transpose_to_bf16_kernel<<<gHH, tb, 0, stream>>>(L[l].wv, Wqkv[l] + 2 * HDIM * HDIM, HDIM, HDIM);
    transpose_to_bf16_kernel<<<gHH, tb, 0, stream>>>(L[l].wo, Wo[l], HDIM, HDIM);
    transpose_to_bf16_kernel<<<dim3(FF / 32, HDIM / 32), tb, 0, stream>>>(L[l].ff1, Wf1[l], HDIM, FF);
    transpose_to_bf16_kernel<<<dim3(HDIM / 32, FF / 32), tb, 0, stream>>>(L[l].ff2, Wf2[l], FF, HDIM);
    hipMemcpyAsync(Bqkv[l] + 0 * HDIM, L[l].bq, HDIM * 4, hipMemcpyDeviceToDevice, stream);
    hipMemcpyAsync(Bqkv[l] + 1 * HDIM, L[l].bk, HDIM * 4, hipMemcpyDeviceToDevice, stream);
    hipMemcpyAsync(Bqkv[l] + 2 * HDIM, L[l].bv, HDIM * 4, hipMemcpyDeviceToDevice, stream);
  }

  embed_kernel<<<(64 * HDIM + 255) / 256, 256, 0, stream>>>(board, embW, embB, emb);
  hipMemcpyAsync(token, rtok, (size_t)HDIM * 4, hipMemcpyDeviceToDevice, stream);

  const int GB = 64;   // 2 waves: 1 column tile, split-K x2

  for (int step = 0; step < MAX_STEPS; ++step) {
    concat_tokens_kernel<<<(SEQ * HDIM + 255) / 256, 256, 0, stream>>>(emb, token, xf, xb);
    for (int l = 0; l < NL; ++l) {
      // fused QKV projection: [65x1024] @ [1024x3072]
      wmma_gemm_kernel<false, false><<<QKVN / 16, GB, 0, stream>>>(
          xb, Wqkv[l], Bqkv[l], nullptr, nullptr, yqkv, nullptr, QKVN, HDIM, SEQ);
      attention_kernel<<<NH, 256, 0, stream>>>(yqkv, ob);
      // output projection + residual, then LN1 (+ per-row std)
      wmma_gemm_kernel<false, true><<<HDIM / 16, GB, 0, stream>>>(
          ob, Wo[l], L[l].bo, xf, nullptr, tmp, nullptr, HDIM, HDIM, SEQ);
      layernorm_kernel<true><<<SEQ, 256, 0, stream>>>(tmp, L[l].ln1s, L[l].ln1b, xf, xb, stats);
      // FFN: relu + adaptive-width mask (confusion folded into epilogue)
      wmma_gemm_kernel<true, false><<<FF / 16, GB, 0, stream>>>(
          xb, Wf1[l], L[l].b1, nullptr, stats, nullptr, hb, FF, HDIM, SEQ);
      wmma_gemm_kernel<false, true><<<HDIM / 16, GB, 0, stream>>>(
          hb, Wf2[l], L[l].b2, xf, nullptr, tmp, nullptr, HDIM, FF, SEQ);
      layernorm_kernel<false><<<SEQ, 256, 0, stream>>>(tmp, L[l].ln2s, L[l].ln2b, xf, xb, nullptr);
    }
    gate_kernel<<<1, 256, 0, stream>>>(xf, token, dw, db, sw, sb, st);
  }

  // heads
  board_mean_kernel<<<(HDIM + 255) / 256, 256, 0, stream>>>(emb, bm);
  head_logits_kernel<<<(PDIM + 255) / 256, 256, 0, stream>>>(
      token, bm, pw, pb, cw, cb, logits, legal);
  final_kernel<<<1, 512, 0, stream>>>(token, vw, vb, logits, legal, st, (float*)d_out);
}
